// EnhancedLIFWithMemory_57535381897774
// MI455X (gfx1250) — matrix operand order repaired and checked by reading, not compiled, not run on hardware
//
#include <hip/hip_runtime.h>
#include <hip/hip_bf16.h>

// Fused LIF-SNN kernel for MI455X (gfx1250, wave32).
//  - GEMM via v_wmma_f32_16x16x32_f16 (f16 in / f32 acc): spikes in [0,1],
//    W ~ N(0, 1/16) -> f16 rel err ~2^-11; f16 WMMA rate >> f32 16x16x4 rate.
//  - W fragments converted once and held in 128 VGPRs per wave for the whole
//    kernel (zero per-tile W traffic).
//  - A (spike) tile double-buffered in LDS and software-pipelined: global
//    loads for tile t+1 are issued before the WMMAs of tile t, converted and
//    stored to the alternate buffer afterwards -> HBM latency hidden, and the
//    "A ready" barrier is eliminated (3 barriers/tile).
//  - Scan is strictly serial in T per feature -> grid = 64 WGs (one/batch),
//    512 threads (16 waves). LN over F handled in-workgroup (wave<->timestep)
//    so currents/spikes never round-trip HBM: ~200MB total HBM traffic
//    (~8.6us floor at 23.3 TB/s).
//  - Reference constants in f32: exp(-200) underflows to 0.0f (as in JAX f32).

typedef __attribute__((ext_vector_type(16))) _Float16 v16h;
typedef __attribute__((ext_vector_type(8)))  float    v8f;
typedef __attribute__((ext_vector_type(4)))  float    v4f;
typedef __attribute__((ext_vector_type(4)))  unsigned int v4u;
typedef __attribute__((ext_vector_type(2)))  unsigned int v2u;

#define T_DIM 1024
#define D_IN  256
#define F_DIM 512
#define TT    16            // timesteps per tile (WMMA M)
#define NTILES (T_DIM / TT) // 64
#define A_STRIDE 132        // dwords per A row (128 pairs + pad, mult of 4 for b128)
#define C_STRIDE 513        // floats per currents row (+1 pad)

#define ALPHA_SYN   0.0f           // expf(-1/0.005) = e^-200 underflows f32
#define ALPHA_MEM   1.9287498e-22f // e^-50
#define ALPHA_ADAPT 4.5399931e-05f // e^-10

union Frag16 { v16h v; v4u q[2]; unsigned int u[8]; _Float16 h[16]; };
union Acc8   { v8f v; float f[8]; };
union Pack4  { _Float16 h[4]; v2u u2; };

__launch_bounds__(512)
__global__ void lif_fused_kernel(const float* __restrict__ spikes,
                                 const float* __restrict__ W_in,
                                 const float* __restrict__ b_in,
                                 const float* __restrict__ thr_ad,
                                 const float* __restrict__ mem_w,
                                 const float* __restrict__ ln_scale,
                                 const float* __restrict__ ln_bias,
                                 float* __restrict__ out)
{
    __shared__ unsigned int Alds[2][TT * A_STRIDE]; // 2 x 8448 B: A tiles (f16 pairs)
    __shared__ float        cur [TT * C_STRIDE];    // 32832 B: currents -> spikes (in place)
    __shared__ float        gammaS[F_DIM];
    __shared__ float        betaS [F_DIM];

    const int tid  = threadIdx.x;
    const int lane = tid & 31;
    const int wv   = tid >> 5;          // 16 waves
    const int b    = blockIdx.x;        // batch
    const int hh   = lane >> 4;         // lane half (WMMA layout)
    const int nl   = lane & 15;         // N within 16x16 tile
    const int fA   = wv * 32 + nl;      // this wave's N-tile 0 feature
    const int fB   = fA + 16;           // N-tile 1 feature

    // Stage LN affine params once.
    gammaS[tid] = ln_scale[tid];
    betaS[tid]  = ln_bias[tid];

    // Per-feature scan constants (thread <-> feature).
    const float bias = b_in[tid];
    const float mw   = mem_w[tid];
    const float thr  = 0.5f + thr_ad[tid];

    // ---- B (weight) fragments, resident in VGPRs for the whole kernel ----
    // B 32x16 f16 layout: lanes 0-15 hold K=0..7,16..23 (pairs per VGPR),
    // lanes 16-31 hold K=8..15,24..31; N = lane & 15.
    Frag16 Bf[2][8];
    #pragma unroll
    for (int kk = 0; kk < 8; ++kk) {
        #pragma unroll
        for (int v = 0; v < 8; ++v) {
            const int k0 = kk * 32 + hh * 8 + 2 * v + (v >= 4 ? 8 : 0);
            Bf[0][kk].h[2*v]   = (_Float16)W_in[(size_t)k0       * F_DIM + fA];
            Bf[0][kk].h[2*v+1] = (_Float16)W_in[(size_t)(k0 + 1) * F_DIM + fA];
            Bf[1][kk].h[2*v]   = (_Float16)W_in[(size_t)k0       * F_DIM + fB];
            Bf[1][kk].h[2*v+1] = (_Float16)W_in[(size_t)(k0 + 1) * F_DIM + fB];
        }
    }

    // Persistent LIF state for feature `tid` across all of T.
    float v_mem = 0.0f, a_ad = 0.0f, m_mem = 0.0f;

    const float* spk_b = spikes + (size_t)b * T_DIM * D_IN;
    float*       out_b = out    + (size_t)b * T_DIM * F_DIM;

    // Per-thread staging geometry: 1024 float4-quads per 16x256 tile, 2/thread.
    // quad q of row m covers K = 4q..4q+3 -> pairs 2q, 2q+1 (contiguous dwords).
    const int sm0 = tid >> 6;              // row of quad 0 (i=0)
    const int sq0 = tid & 63;              // quad-in-row
    const int sm1 = (tid + 512) >> 6;      // row of quad 1 (i=1)
    const int sq1 = (tid + 512) & 63;

    // ---- Prologue: stage tile 0 into buffer 0 ----
    {
        const v4f r0 = *(const v4f*)(spk_b + (size_t)sm0 * D_IN + 4 * sq0);
        const v4f r1 = *(const v4f*)(spk_b + (size_t)sm1 * D_IN + 4 * sq1);
        Pack4 p0, p1;
        p0.h[0] = (_Float16)r0[0]; p0.h[1] = (_Float16)r0[1];
        p0.h[2] = (_Float16)r0[2]; p0.h[3] = (_Float16)r0[3];
        p1.h[0] = (_Float16)r1[0]; p1.h[1] = (_Float16)r1[1];
        p1.h[2] = (_Float16)r1[2]; p1.h[3] = (_Float16)r1[3];
        *(v2u*)&Alds[0][sm0 * A_STRIDE + 2 * sq0] = p0.u2;
        *(v2u*)&Alds[0][sm1 * A_STRIDE + 2 * sq1] = p1.u2;
    }
    __syncthreads();   // tile-0 A visible to all waves

    for (int tile = 0; tile < NTILES; ++tile) {
        const int buf  = tile & 1;
        const bool nxt = (tile + 1) < NTILES;

        // ---- Issue global loads for tile+1 (latency hidden behind WMMAs) ----
        v4f r0 = {}, r1 = {};
        if (nxt) {
            const float* base = spk_b + (size_t)(tile + 1) * TT * D_IN;
            const float* s0 = base + (size_t)sm0 * D_IN + 4 * sq0;
            const float* s1 = base + (size_t)sm1 * D_IN + 4 * sq1;
            r0 = *(const v4f*)s0;
            r1 = *(const v4f*)s1;
            if (tile + 2 < NTILES) {                    // -> global_prefetch_b8
                __builtin_prefetch(s0 + TT * D_IN, 0, 1);
                __builtin_prefetch(s1 + TT * D_IN, 0, 1);
            }
        }

        // ---- GEMM: currents[16 x 512] = A(16x256) x W(256x512), 32 feats/wave ----
        Acc8 acc0 = {}, acc1 = {};
        const unsigned int* arow = &Alds[buf][nl * A_STRIDE + 4 * hh];
        #pragma unroll
        for (int kk = 0; kk < 8; ++kk) {
            Frag16 af;                                     // A 16x32 f16 fragment
            af.q[0] = *(const v4u*)(arow + kk * 16);       // pairs 4h+0..3   (VGPR 0-3)
            af.q[1] = *(const v4u*)(arow + kk * 16 + 8);   // pairs 4h+8..11  (VGPR 4-7)
            acc0.v = __builtin_amdgcn_wmma_f32_16x16x32_f16(
                         false, af.v, false, Bf[0][kk].v, (short)0, acc0.v, false, false);
            acc1.v = __builtin_amdgcn_wmma_f32_16x16x32_f16(
                         false, af.v, false, Bf[1][kk].v, (short)0, acc1.v, false, false);
        }

        // ---- Convert + store tile+1 A into the alternate buffer ----
        if (nxt) {
            Pack4 p0, p1;
            p0.h[0] = (_Float16)r0[0]; p0.h[1] = (_Float16)r0[1];
            p0.h[2] = (_Float16)r0[2]; p0.h[3] = (_Float16)r0[3];
            p1.h[0] = (_Float16)r1[0]; p1.h[1] = (_Float16)r1[1];
            p1.h[2] = (_Float16)r1[2]; p1.h[3] = (_Float16)r1[3];
            *(v2u*)&Alds[buf ^ 1][sm0 * A_STRIDE + 2 * sq0] = p0.u2;
            *(v2u*)&Alds[buf ^ 1][sm1 * A_STRIDE + 2 * sq1] = p1.u2;
        }

        // C/D layout: VGPR r, lane l -> M = r + 8*(l>>4), N = l&15.
        #pragma unroll
        for (int r = 0; r < 8; ++r) {
            const int row = r + 8 * hh;
            cur[row * C_STRIDE + fA] = acc0.f[r];
            cur[row * C_STRIDE + fB] = acc1.f[r];
        }
        __syncthreads();   // currents ready (A[buf^1] stores also ordered here)

        // ---- LIF scan, serial over the 16 timesteps (thread <-> feature) ----
        #pragma unroll
        for (int t = 0; t < TT; ++t) {
            const float x     = cur[t * C_STRIDE + tid] + bias;
            const float total = ALPHA_SYN * x + mw * m_mem;
            const float vv    = ALPHA_MEM * v_mem + (1.0f - ALPHA_MEM) * total;
            const float s     = (vv - thr) > 0.0f ? 1.0f : 0.0f;
            a_ad  = ALPHA_ADAPT * a_ad + (1.0f - ALPHA_ADAPT) * s * 0.01f;
            v_mem = vv * (1.0f - s) + (0.0f - a_ad) * s;
            m_mem = 0.95f * m_mem + 0.05f * s;
            cur[t * C_STRIDE + tid] = s;           // in place: only this thread touches [t][tid]
        }
        __syncthreads();   // spikes ready

        // ---- LayerNorm: wave wv <-> timestep t0+wv; s in {0,1} => var = mu - mu^2 ----
        float vals[16];
        float sum = 0.0f;
        #pragma unroll
        for (int i = 0; i < 16; ++i) {
            vals[i] = cur[wv * C_STRIDE + i * 32 + lane];
            sum += vals[i];
        }
        #pragma unroll
        for (int off = 16; off > 0; off >>= 1)
            sum += __shfl_xor(sum, off, 32);
        const float mean = sum * (1.0f / (float)F_DIM);
        const float rstd = rsqrtf(fmaxf(mean - mean * mean, 0.0f) + 1e-6f);
        float* orow = out_b + (size_t)(tile * TT + wv) * F_DIM;
        #pragma unroll
        for (int i = 0; i < 16; ++i) {
            const int f = i * 32 + lane;
            orow[f] = (vals[i] - mean) * rstd * gammaS[f] + betaS[f];
        }
        __syncthreads();   // cur free for next tile's accumulator stores
    }
}

extern "C" void kernel_launch(void* const* d_in, const int* in_sizes, int n_in,
                              void* d_out, int out_size, void* d_ws, size_t ws_size,
                              hipStream_t stream) {
    (void)in_sizes; (void)n_in; (void)out_size; (void)d_ws; (void)ws_size;
    const float* spikes   = (const float*)d_in[0];
    const float* W_in     = (const float*)d_in[1];
    const float* b_in     = (const float*)d_in[2];
    const float* thr_ad   = (const float*)d_in[3];
    const float* mem_w    = (const float*)d_in[4];
    const float* ln_scale = (const float*)d_in[5];
    const float* ln_bias  = (const float*)d_in[6];
    float* out = (float*)d_out;

    lif_fused_kernel<<<dim3(64), dim3(512), 0, stream>>>(
        spikes, W_in, b_in, thr_ad, mem_w, ln_scale, ln_bias, out);
}
